// Self_Attention_91104846283288
// MI455X (gfx1250) — matrix operand order repaired and checked by reading, not compiled, not run on hardware
//
#include <hip/hip_runtime.h>
#include <hip/hip_bf16.h>

typedef __attribute__((ext_vector_type(16))) __bf16 v16bf;
typedef __attribute__((ext_vector_type(8)))  __bf16 v8bf;
typedef __attribute__((ext_vector_type(8)))  float  v8f;
typedef int i32x4 __attribute__((vector_size(16)));

union BF16x16 { v16bf v; v8bf h[2]; };

// Async global->LDS path (CDNA5 GLOBAL_LOAD_ASYNC_TO_LDS_B128 / ASYNCcnt),
// guarded so the kernel still compiles if the builtins are absent.
#if defined(__gfx1250__) && __has_builtin(__builtin_amdgcn_global_load_async_to_lds_b128) && __has_builtin(__builtin_amdgcn_s_wait_asynccnt)
#define USE_ASYNC_LDS 1
#else
#define USE_ASYNC_LDS 0
#endif

__device__ inline void copy16B(const __bf16* g, __bf16* l) {
#if USE_ASYNC_LDS
  __builtin_amdgcn_global_load_async_to_lds_b128(
      (__attribute__((address_space(1))) i32x4*)(void*)g,
      (__attribute__((address_space(3))) i32x4*)(void*)l, 0, 0);
#else
  *(v8bf*)l = *(const v8bf*)g;
#endif
}

__device__ inline void async_fence() {
#if USE_ASYNC_LDS
  __builtin_amdgcn_s_wait_asynccnt(0);
#endif
}

__device__ inline v8f wmma_bf16(v16bf a, v16bf b, v8f c) {
  // D = A(16x32 bf16) * B(32x16 bf16) + C(16x16 f32)
  return __builtin_amdgcn_wmma_f32_16x16x32_bf16(false, a, false, b,
                                                 (short)0, c, false, false);
}

__device__ inline v8f zero8() {
  v8f z;
#pragma unroll
  for (int i = 0; i < 8; ++i) z[i] = 0.0f;
  return z;
}

// ---------------------------------------------------------------- convert
__global__ __launch_bounds__(256) void f32_to_bf16_kernel(
    const float* __restrict__ in, __bf16* __restrict__ out, int n) {
  for (int i = blockIdx.x * blockDim.x + threadIdx.x; i < n;
       i += gridDim.x * blockDim.x)
    out[i] = (__bf16)in[i];
}

// ------------------------------------------------------- shared GEMM pieces
// Stage a 128x32 bf16 tile (row stride 1024 in global, 40 in LDS).
__device__ inline void stage_tile(const __bf16* __restrict__ G, int row0,
                                  int k0, __bf16* dst, int tid) {
#pragma unroll
  for (int i = 0; i < 2; ++i) {
    int c = tid + i * 256, row = c >> 2, ch = c & 3;
    copy16B(G + (size_t)(row0 + row) * 1024 + k0 + ch * 8,
            dst + row * 40 + ch * 8);
  }
}

// Double-buffered 128x128x1024 mainloop; each wave produces a 32x64 tile
// as acc[2][4] (8 WMMAs per 32-deep k-step).
__device__ inline void gemm_mainloop(const __bf16* __restrict__ A,
                                     const __bf16* __restrict__ W, int m0,
                                     int n0, __bf16 (*sm)[128 * 40],
                                     v8f acc[2][4], int tid, int wm, int wn,
                                     int ln, int hi) {
  stage_tile(A, m0, 0, sm[0], tid);
  stage_tile(W, n0, 0, sm[1], tid);
  async_fence();
  __syncthreads();
  for (int ks = 0; ks < 32; ++ks) {
    const int buf = ks & 1;
    if (ks + 1 < 32) {  // prefetch next k-slice into the other buffer
      stage_tile(A, m0, (ks + 1) * 32, sm[(buf ^ 1) * 2 + 0], tid);
      stage_tile(W, n0, (ks + 1) * 32, sm[(buf ^ 1) * 2 + 1], tid);
    }
    const __bf16* As = sm[buf * 2 + 0];
    const __bf16* Bs = sm[buf * 2 + 1];
    BF16x16 a[2], b[4];
#pragma unroll
    for (int i = 0; i < 2; ++i) {
      const __bf16* ap = &As[(wm * 32 + i * 16 + ln) * 40 + hi * 8];
      a[i].h[0] = *(const v8bf*)ap;
      a[i].h[1] = *(const v8bf*)(ap + 16);
    }
#pragma unroll
    for (int nt = 0; nt < 4; ++nt) {
      const __bf16* bp = &Bs[(wn * 64 + nt * 16 + ln) * 40 + hi * 8];
      b[nt].h[0] = *(const v8bf*)bp;
      b[nt].h[1] = *(const v8bf*)(bp + 16);
    }
#pragma unroll
    for (int i = 0; i < 2; ++i)
#pragma unroll
      for (int nt = 0; nt < 4; ++nt)
        acc[i][nt] = wmma_bf16(a[i].v, b[nt].v, acc[i][nt]);
    async_fence();
    __syncthreads();
  }
}

// ---------------------------------------------------------------- QKV GEMM
// C[4096,3072] = Xh[4096,1024] * Win[3072,1024]^T + b_in, scattered into
// per-head Q/K/V [B,H,S,64] bf16 buffers.
__global__ __launch_bounds__(256) void qkv_gemm_kernel(
    const __bf16* __restrict__ A, const __bf16* __restrict__ W,
    const float* __restrict__ bias, __bf16* __restrict__ Q,
    __bf16* __restrict__ Kc, __bf16* __restrict__ V) {
  __shared__ __align__(16) __bf16 sm[4][128 * 40];  // [buf*2 + (A/B)]
  const int tid = threadIdx.x, wave = tid >> 5, lane = tid & 31;
  const int ln = lane & 15, hi = lane >> 4;
  const int wm = wave & 3, wn = wave >> 2;
  const int m0 = blockIdx.x * 128, n0 = blockIdx.y * 128;

  v8f acc[2][4];
#pragma unroll
  for (int i = 0; i < 2; ++i)
#pragma unroll
    for (int nt = 0; nt < 4; ++nt) acc[i][nt] = zero8();

  gemm_mainloop(A, W, m0, n0, sm, acc, tid, wm, wn, ln, hi);

  // epilogue: bias + scatter into [B,H,S,64]
#pragma unroll
  for (int nt = 0; nt < 4; ++nt) {
    int n = n0 + wn * 64 + nt * 16 + ln;
    float bv = bias[n];
    int comp = n >> 10, d = n & 1023, head = d >> 6, e = d & 63;
    __bf16* dstbase = (comp == 0) ? Q : ((comp == 1) ? Kc : V);
#pragma unroll
    for (int i = 0; i < 2; ++i)
#pragma unroll
      for (int r = 0; r < 8; ++r) {
        int m = m0 + wm * 32 + i * 16 + r + hi * 8;
        int bidx = m >> 11, s = m & 2047;
        dstbase[(((size_t)(bidx * 16 + head)) * 2048 + s) * 64 + e] =
            (__bf16)(acc[i][nt][r] + bv);
      }
  }
}

// ---------------------------------------------------------------- out proj
// out[4096,1024] = Ah[4096,1024] * Wout[1024,1024]^T + b_out (f32 result)
__global__ __launch_bounds__(256) void out_gemm_kernel(
    const __bf16* __restrict__ A, const __bf16* __restrict__ W,
    const float* __restrict__ bias, float* __restrict__ out) {
  __shared__ __align__(16) __bf16 sm[4][128 * 40];
  const int tid = threadIdx.x, wave = tid >> 5, lane = tid & 31;
  const int ln = lane & 15, hi = lane >> 4;
  const int wm = wave & 3, wn = wave >> 2;
  const int m0 = blockIdx.x * 128, n0 = blockIdx.y * 128;

  v8f acc[2][4];
#pragma unroll
  for (int i = 0; i < 2; ++i)
#pragma unroll
    for (int nt = 0; nt < 4; ++nt) acc[i][nt] = zero8();

  gemm_mainloop(A, W, m0, n0, sm, acc, tid, wm, wn, ln, hi);

#pragma unroll
  for (int nt = 0; nt < 4; ++nt) {
    int n = n0 + wn * 64 + nt * 16 + ln;
    float bv = bias[n];
#pragma unroll
    for (int i = 0; i < 2; ++i)
#pragma unroll
      for (int r = 0; r < 8; ++r) {
        int m = m0 + wm * 32 + i * 16 + r + hi * 8;
        out[(size_t)m * 1024 + n] = acc[i][nt][r] + bv;
      }
  }
}

// ---------------------------------------------------------------- attention
// Flash attention per (b, head, 128-row q tile); each wave owns 16 q rows.
__global__ __launch_bounds__(256) void attn_kernel(
    const __bf16* __restrict__ Qh, const __bf16* __restrict__ Kh,
    const __bf16* __restrict__ Vh, __bf16* __restrict__ Oh,
    const int* __restrict__ causal) {
  __shared__ __align__(16) __bf16 Ks[64 * 72];      // K tile  [k][d]
  __shared__ __align__(16) __bf16 Vt[64 * 72];      // V tile transposed [e][k]
  __shared__ __align__(16) __bf16 Ps[8 * 16 * 72];  // per-wave P [q][k]
  const int tid = threadIdx.x, wave = tid >> 5, lane = tid & 31;
  const int ln = lane & 15, hi = lane >> 4;
  const int q0 = blockIdx.x * 128, head = blockIdx.y, b = blockIdx.z;
  const size_t hb = (size_t)(b * 16 + head) * 2048;
  const __bf16* Qp = Qh + hb * 64;
  const __bf16* Kp = Kh + hb * 64;
  const __bf16* Vp = Vh + hb * 64;
  const bool cm = (causal[0] != 0);

  // Q fragments for this wave's 16 rows (A-operand layout), hoisted.
  BF16x16 qa[2];
  {
    const __bf16* qptr = Qp + (size_t)(q0 + wave * 16 + ln) * 64;
#pragma unroll
    for (int ks = 0; ks < 2; ++ks) {
      qa[ks].h[0] = *(const v8bf*)(qptr + ks * 32 + hi * 8);
      qa[ks].h[1] = *(const v8bf*)(qptr + ks * 32 + hi * 8 + 16);
    }
  }

  v8f o[4];
#pragma unroll
  for (int nt = 0; nt < 4; ++nt) o[nt] = zero8();
  float mrun[8], lrun[8];
#pragma unroll
  for (int r = 0; r < 8; ++r) { mrun[r] = -3.0e38f; lrun[r] = 0.0f; }

  const int ntiles = cm ? (q0 / 64 + 2) : (2048 / 64);
  __bf16* pw = Ps + wave * 16 * 72;

  for (int t = 0; t < ntiles; ++t) {
    const int kv0 = t * 64;
    __syncthreads();
#pragma unroll
    for (int i = 0; i < 2; ++i) {  // stage K tile 64x64 row-major (async)
      int c = tid + i * 256, row = c >> 3, off = (c & 7) * 8;
      copy16B(Kp + (size_t)(kv0 + row) * 64 + off, &Ks[row * 72 + off]);
    }
    {  // stage V tile transposed: Vt[e][k]
      int k = tid & 63, eg = (tid >> 6) * 16;
      const __bf16* vp = Vp + (size_t)(kv0 + k) * 64 + eg;
      v8bf v0 = *(const v8bf*)vp, v1 = *(const v8bf*)(vp + 8);
#pragma unroll
      for (int j = 0; j < 8; ++j) {
        Vt[(eg + j) * 72 + k] = v0[j];
        Vt[(eg + 8 + j) * 72 + k] = v1[j];
      }
    }
    if (t + 1 < ntiles) {  // prefetch next KV tile into cache
      __builtin_prefetch(Kp + (size_t)(kv0 + 64) * 64 + tid * 16, 0, 1);
      __builtin_prefetch(Vp + (size_t)(kv0 + 64) * 64 + tid * 16, 0, 1);
    }
    async_fence();
    __syncthreads();

    // S = Q * K^T  (16x64 per wave, 8 WMMAs); fetch all B frags first
    v8f sc[4];
#pragma unroll
    for (int nt = 0; nt < 4; ++nt) sc[nt] = zero8();
#pragma unroll
    for (int ks = 0; ks < 2; ++ks) {
      BF16x16 bm[4];
#pragma unroll
      for (int nt = 0; nt < 4; ++nt) {
        const __bf16* kp = &Ks[(nt * 16 + ln) * 72 + ks * 32 + hi * 8];
        bm[nt].h[0] = *(const v8bf*)kp;
        bm[nt].h[1] = *(const v8bf*)(kp + 16);
      }
#pragma unroll
      for (int nt = 0; nt < 4; ++nt)
        sc[nt] = wmma_bf16(qa[ks].v, bm[nt].v, sc[nt]);
    }

    // online softmax (scale = hd^-0.5 = 0.125)
#pragma unroll
    for (int r = 0; r < 8; ++r) {
      const int row = q0 + wave * 16 + r + hi * 8;
      float mx = -3.0e38f;
#pragma unroll
      for (int nt = 0; nt < 4; ++nt) {
        int col = kv0 + nt * 16 + ln;
        float s = sc[nt][r] * 0.125f;
        if (cm && col > row) s = -1.0e9f;
        sc[nt][r] = s;
        mx = fmaxf(mx, s);
      }
      for (int msk = 1; msk < 16; msk <<= 1)
        mx = fmaxf(mx, __shfl_xor(mx, msk, 32));
      float mn = fmaxf(mrun[r], mx);
      float al = __expf(mrun[r] - mn);
      mrun[r] = mn;
      float rs = 0.0f;
#pragma unroll
      for (int nt = 0; nt < 4; ++nt) {
        float p = __expf(sc[nt][r] - mn);
        sc[nt][r] = p;
        rs += p;
      }
      for (int msk = 1; msk < 16; msk <<= 1) rs += __shfl_xor(rs, msk, 32);
      lrun[r] = lrun[r] * al + rs;
#pragma unroll
      for (int nt = 0; nt < 4; ++nt) o[nt][r] *= al;
    }

    // P (C layout) -> per-wave LDS, then O += P * V
#pragma unroll
    for (int nt = 0; nt < 4; ++nt)
#pragma unroll
      for (int r = 0; r < 8; ++r)
        pw[(r + hi * 8) * 72 + nt * 16 + ln] = (__bf16)sc[nt][r];
#pragma unroll
    for (int ks = 0; ks < 2; ++ks) {
      BF16x16 pa;
      const __bf16* pp = pw + ln * 72 + ks * 32 + hi * 8;
      pa.h[0] = *(const v8bf*)pp;
      pa.h[1] = *(const v8bf*)(pp + 16);
      BF16x16 bm[4];
#pragma unroll
      for (int nt = 0; nt < 4; ++nt) {
        const __bf16* vp = &Vt[(nt * 16 + ln) * 72 + ks * 32 + hi * 8];
        bm[nt].h[0] = *(const v8bf*)vp;
        bm[nt].h[1] = *(const v8bf*)(vp + 16);
      }
#pragma unroll
      for (int nt = 0; nt < 4; ++nt)
        o[nt] = wmma_bf16(pa.v, bm[nt].v, o[nt]);
    }
  }

  // epilogue: O / l -> [B,S,D] bf16
#pragma unroll
  for (int r = 0; r < 8; ++r) {
    float inv = 1.0f / lrun[r];
    int s = q0 + wave * 16 + r + hi * 8;
    __bf16* dst = Oh + ((size_t)(b * 2048 + s)) * 1024 + head * 64;
#pragma unroll
    for (int nt = 0; nt < 4; ++nt)
      dst[nt * 16 + ln] = (__bf16)(o[nt][r] * inv);
  }
}

// ---------------------------------------------------------------- launch
extern "C" void kernel_launch(void* const* d_in, const int* in_sizes, int n_in,
                              void* d_out, int out_size, void* d_ws,
                              size_t ws_size, hipStream_t stream) {
  const float* x = (const float*)d_in[0];      // [2,2048,1024]
  const float* w_in = (const float*)d_in[1];   // [3072,1024]
  const float* b_in = (const float*)d_in[2];   // [3072]
  const float* w_out = (const float*)d_in[3];  // [1024,1024]
  const float* b_out = (const float*)d_in[4];  // [1024]
  const int* causal = (const int*)d_in[5];     // scalar

  char* ws = (char*)d_ws;
  __bf16* xh    = (__bf16*)(ws);                        // 8 MB
  __bf16* winh  = (__bf16*)(ws + (8u << 20));           // 6 MB
  __bf16* wouth = (__bf16*)(ws + (14u << 20));          // 2 MB
  __bf16* Qh    = (__bf16*)(ws + (16u << 20));          // 8 MB [B,H,S,64]
  __bf16* Kh    = (__bf16*)(ws + (24u << 20));          // 8 MB
  __bf16* Vh    = (__bf16*)(ws + (32u << 20));          // 8 MB
  __bf16* Ah    = (__bf16*)(ws + (40u << 20));          // 8 MB [B,S,D]

  f32_to_bf16_kernel<<<1024, 256, 0, stream>>>(x, xh, 4096 * 1024);
  f32_to_bf16_kernel<<<1024, 256, 0, stream>>>(w_in, winh, 3072 * 1024);
  f32_to_bf16_kernel<<<1024, 256, 0, stream>>>(w_out, wouth, 1024 * 1024);

  qkv_gemm_kernel<<<dim3(32, 24), 256, 0, stream>>>(xh, winh, b_in, Qh, Kh, Vh);
  attn_kernel<<<dim3(16, 16, 2), 256, 0, stream>>>(Qh, Kh, Vh, Ah, causal);
  out_gemm_kernel<<<dim3(32, 8), 256, 0, stream>>>(Ah, wouth, b_out,
                                                   (float*)d_out);
}
